// CPN_18408229830738
// MI455X (gfx1250) — compile-verified, matrix operand order checked by reading, not run on hardware
//
#include <hip/hip_runtime.h>
#include <cstdint>
#include <cstddef>

// =====================================================================
//  MI455X (gfx1250) implementation of the point-cloud completion net.
//  All contractions (3D convs via im2col, pointwise convs, decoder
//  einsums, MLPs) go through v_wmma_f32_16x16x32_f16 (f16 in, f32 acc).
//  GEMM staging: A tile via async global->LDS (b128) when in-bounds,
//  B tile stored LDS-pre-swizzled into the WMMA fragment layout so each
//  lane reads its fragment with two ds_load_b128.
//  Sequential MST / min-density-sampling run as one-WG LDS-resident
//  kernels (wave32 tree reductions).
//
//  Assumptions (compile-only environment, documented):
//   * d_in follows JAX pytree flattening order (dict keys sorted).
//   * d_ws is large enough (~250 MB) for the carved scratch below.
// =====================================================================

typedef _Float16 f16;
typedef __attribute__((ext_vector_type(16))) _Float16 v16h;
typedef __attribute__((ext_vector_type(8)))  _Float16 v8h;
typedef __attribute__((ext_vector_type(8)))  float    v8f;

#define ACT_NONE  0
#define ACT_RELU  1
#define ACT_LEAKY 2
#define ACT_ELU   3

static constexpr int   kB      = 4;
static constexpr int   kN      = 2048;    // points per cloud
static constexpr int   kP      = 8;
static constexpr int   kG      = 128;
static constexpr int   kLatent = 1024;
static constexpr int   kNumCat = 48;
static constexpr int   kMdsM   = 3072;    // N_IN + P*G
static constexpr int   kMdsN   = 2048;
static constexpr float kAlpha  = 1.5f;

// ---------------------------------------------------------------------
//  WMMA GEMM:  C[M,N] = act(A[M,Kpad] * B[K,N] + bias[N]) (f16 in, f32 out)
//  A rows padded: lda = 32-multiple, 16B aligned, pad region zeroed.
//  Block tile 128x64, 8 waves (4x2), each wave 32x32 = 2x2 WMMA tiles.
// ---------------------------------------------------------------------
__device__ __forceinline__ float act_apply(float x, int act) {
  if (act == ACT_RELU)  return x > 0.f ? x : 0.f;
  if (act == ACT_LEAKY) return x > 0.f ? x : 0.1f * x;
  if (act == ACT_ELU)   return x > 0.f ? x : (expf(x) - 1.f);
  return x;
}

__global__ void __launch_bounds__(256)
k_gemm_wmma(const f16* __restrict__ A, const f16* __restrict__ Bm,
            float* __restrict__ C, const float* __restrict__ bias,
            int M, int N, int K, int lda, int ldc, int act)
{
  // A tile: natural row-major (row stride 40 halves = 80B, 16B aligned).
  __shared__ __align__(16) f16 As[128][40];
  // B tile pre-swizzled to fragment layout: [n_subtile][lane][16 halves]
  // lane = ((k>>3)&1)*16 + (col&15); half = (k>>4)*8 + (k&7).
  __shared__ __align__(16) f16 Bs[4][32][24];   // padded to 48B/lane

  const int tid    = threadIdx.x;
  const int lane   = tid & 31;
  const int wave   = tid >> 5;
  const int wm     = wave >> 1;     // 0..3
  const int wn     = wave & 1;      // 0..1
  const int row_hi = lane >> 4;
  const int row_lo = lane & 15;

  const int bm = blockIdx.y * 128;
  const int bn = blockIdx.x * 64;
  const bool a_full = (bm + 128 <= M);   // uniform per block

  union CF { v8f v; float f[8]; };
  CF acc[2][2];
  for (int i = 0; i < 2; ++i)
    for (int j = 0; j < 2; ++j)
      for (int v = 0; v < 8; ++v) acc[i][j].f[v] = 0.f;

  for (int k0 = 0; k0 < K; k0 += 32) {
    if (k0 + 32 < K)   // warm GL2 for next A tile
      __builtin_prefetch(A + (size_t)(bm + (tid >> 1)) * lda + k0 + 32, 0, 1);

    // ---- stage A tile: 128 rows x 32 halves, 16B vector groups ----
    if (a_full) {
      // CDNA5 async global->LDS copy (ASYNCcnt-tracked); lda padding
      // guarantees the 16B group never crosses the padded row end.
#pragma unroll
      for (int ps = 0; ps < 2; ++ps) {
        int g = ps * 256 + tid;
        int r = g >> 2, cg = g & 3;
        unsigned ldsa = (unsigned)(size_t)(void*)&As[r][cg * 8];
        const f16* gp = A + (size_t)(bm + r) * lda + k0 + cg * 8;
        asm volatile("global_load_async_to_lds_b128 %0, %1, off"
                     :: "v"(ldsa), "v"(gp) : "memory");
      }
      asm volatile("s_wait_asynccnt 0x0" ::: "memory");
    } else {
#pragma unroll
      for (int ps = 0; ps < 2; ++ps) {
        int g = ps * 256 + tid;
        int r = g >> 2, cg = g & 3;
        int gm = bm + r;
        union V8 { v8h v; f16 h[8]; } val;
        if (gm < M) {
          val.v = *(const v8h*)(A + (size_t)gm * lda + k0 + cg * 8);
        } else {
#pragma unroll
          for (int e = 0; e < 8; ++e) val.h[e] = (f16)0;
        }
        *(v8h*)&As[r][cg * 8] = val.v;
      }
    }

    // ---- stage B tile swizzled: thread t = (kq = t>>6, col = t&63) ----
    {
      int c  = tid & 63;
      int kq = tid >> 6;             // 0..3, covers k = kq*8 .. kq*8+7
      int gn = bn + c;
      bool colok = gn < N;
      union V8 { v8h v; f16 h[8]; } val;
#pragma unroll
      for (int e = 0; e < 8; ++e) {
        int gk = k0 + kq * 8 + e;
        val.h[e] = (colok && gk < K) ? Bm[(size_t)gk * N + gn] : (f16)0;
      }
      int sub = c >> 4;
      int lne = ((kq & 1) << 4) + (c & 15);
      *(v8h*)&Bs[sub][lne][(kq >> 1) * 8] = val.v;
    }
    __syncthreads();

    // ---- fragments: two ds_load_b128 each ----
    union HF { v16h v; v8h q[2]; };
    HF af[2], bf[2];
#pragma unroll
    for (int i = 0; i < 2; ++i) {
      int arow = wm * 32 + i * 16 + row_lo;
      af[i].q[0] = *(const v8h*)&As[arow][row_hi * 8];
      af[i].q[1] = *(const v8h*)&As[arow][16 + row_hi * 8];
      bf[i].q[0] = *(const v8h*)&Bs[wn * 2 + i][lane][0];
      bf[i].q[1] = *(const v8h*)&Bs[wn * 2 + i][lane][8];
    }
#pragma unroll
    for (int i = 0; i < 2; ++i)
#pragma unroll
      for (int j = 0; j < 2; ++j)
        acc[i][j].v = __builtin_amdgcn_wmma_f32_16x16x32_f16(
            false, af[i].v, false, bf[j].v, (short)0, acc[i][j].v, false, false);
    __syncthreads();
  }

  // Epilogue: C/D layout = lane(N), vgpr + lanehalf*8 (M)
  for (int i = 0; i < 2; ++i) {
    for (int j = 0; j < 2; ++j) {
      int n = bn + wn * 32 + j * 16 + row_lo;
      if (n >= N) continue;
      float bv = bias ? bias[n] : 0.f;
#pragma unroll
      for (int v = 0; v < 8; ++v) {
        int m = bm + wm * 32 + i * 16 + row_hi * 8 + v;
        if (m < M) C[(size_t)m * ldc + n] = act_apply(acc[i][j].f[v] + bv, act);
      }
    }
  }
}

// ---------------------------------------------------------------------
//  Utility kernels
// ---------------------------------------------------------------------
__global__ void k_cast_f32_f16(const float* __restrict__ in, f16* __restrict__ out, size_t n) {
  size_t i = (size_t)blockIdx.x * 256 + threadIdx.x;
  if (i < n) out[i] = (f16)in[i];
}

// cast MxK f32 -> Mxlda f16, zero padding cols [K, lda)
__global__ void k_cast_pad(const float* __restrict__ in, f16* __restrict__ out,
                           int M, int K, int lda) {
  size_t total = (size_t)M * lda;
  size_t i = (size_t)blockIdx.x * 256 + threadIdx.x;
  if (i >= total) return;
  int col = (int)(i % lda);
  int row = (int)(i / lda);
  out[i] = (col < K) ? (f16)in[(size_t)row * K + col] : (f16)0;
}

__global__ void k_cast_transpose_f32_f16(const float* __restrict__ in, f16* __restrict__ out,
                                         int M, int K) {   // in: MxK  -> out: KxM
  int k = blockIdx.x * 16 + threadIdx.x;
  int m = blockIdx.y * 16 + threadIdx.y;
  if (m < M && k < K) out[(size_t)k * M + m] = (f16)in[(size_t)m * K + k];
}

__global__ void k_zero_f32(float* p, size_t n) {
  size_t i = (size_t)blockIdx.x * 256 + threadIdx.x;
  if (i < n) p[i] = 0.f;
}

__global__ void k_copy_f32(const float* __restrict__ in, float* __restrict__ out, size_t n) {
  size_t i = (size_t)blockIdx.x * 256 + threadIdx.x;
  if (i < n) out[i] = in[i];
}

// (B,Nr,C) -> (B,C,Nr)
__global__ void k_transpose_bnc_bcn(const float* __restrict__ in, float* __restrict__ out,
                                    int Nr, int C) {
  size_t total = (size_t)kB * Nr * C;
  size_t i = (size_t)blockIdx.x * 256 + threadIdx.x;
  if (i >= total) return;
  int c = (int)(i % C);
  int n = (int)((i / C) % Nr);
  int b = (int)(i / ((size_t)C * Nr));
  out[((size_t)b * C + c) * Nr + n] = in[i];
}

// ---------------------------------------------------------------------
//  Voxelize / devoxelize
// ---------------------------------------------------------------------
__global__ void k_voxel_stats(const float* __restrict__ coords, int cstride,
                              float* __restrict__ meanv, float* __restrict__ scalev, int Nn) {
  int b = blockIdx.x, t = threadIdx.x;
  __shared__ float red[256];
  __shared__ float smean[3];
  const float* cb = coords + (size_t)b * cstride;
  for (int d = 0; d < 3; ++d) {
    float acc = 0.f;
    for (int n = t; n < Nn; n += 256) acc += cb[d * Nn + n];
    red[t] = acc; __syncthreads();
    for (int off = 128; off; off >>= 1) { if (t < off) red[t] += red[t + off]; __syncthreads(); }
    if (t == 0) smean[d] = red[0] / (float)Nn;
    __syncthreads();
  }
  float mx = 0.f;
  for (int n = t; n < Nn; n += 256) {
    float dx = cb[0 * Nn + n] - smean[0];
    float dy = cb[1 * Nn + n] - smean[1];
    float dz = cb[2 * Nn + n] - smean[2];
    mx = fmaxf(mx, sqrtf(dx * dx + dy * dy + dz * dz));
  }
  red[t] = mx; __syncthreads();
  for (int off = 128; off; off >>= 1) { if (t < off) red[t] = fmaxf(red[t], red[t + off]); __syncthreads(); }
  if (t == 0) {
    scalev[b] = red[0] * 2.f + 1e-6f;
    meanv[b * 3 + 0] = smean[0]; meanv[b * 3 + 1] = smean[1]; meanv[b * 3 + 2] = smean[2];
  }
}

__global__ void k_voxel_coords(const float* __restrict__ coords, int cstride,
                               const float* __restrict__ meanv, const float* __restrict__ scalev,
                               float* __restrict__ nc, int* __restrict__ flat, int Nn, int r) {
  int i = blockIdx.x * 256 + threadIdx.x;
  if (i >= kB * Nn) return;
  int b = i / Nn, n = i % Nn;
  float s = scalev[b];
  int vc[3];
#pragma unroll
  for (int d = 0; d < 3; ++d) {
    float v = ((coords[(size_t)b * cstride + d * Nn + n] - meanv[b * 3 + d]) / s + 0.5f) * (float)r;
    v = fminf(fmaxf(v, 0.f), (float)(r - 1));
    nc[((size_t)b * 3 + d) * Nn + n] = v;
    vc[d] = (int)rintf(v);
  }
  flat[(size_t)b * Nn + n] = (vc[0] * r + vc[1]) * r + vc[2];
}

__global__ void k_voxel_scatter(const float* __restrict__ feats, const int* __restrict__ flat,
                                float* __restrict__ vox, float* __restrict__ cnt,
                                int C, int Nn, int R3) {
  size_t total = (size_t)kB * C * Nn;
  size_t i = (size_t)blockIdx.x * 256 + threadIdx.x;
  if (i >= total) return;
  int n = (int)(i % Nn);
  int c = (int)((i / Nn) % C);
  int b = (int)(i / ((size_t)Nn * C));
  int fl = flat[(size_t)b * Nn + n];
  atomicAdd(&vox[((size_t)b * C + c) * R3 + fl], feats[i]);
  if (c == 0) atomicAdd(&cnt[(size_t)b * R3 + fl], 1.f);
}

__global__ void k_voxel_div(float* __restrict__ vox, const float* __restrict__ cnt, int C, int R3) {
  size_t total = (size_t)kB * C * R3;
  size_t i = (size_t)blockIdx.x * 256 + threadIdx.x;
  if (i >= total) return;
  int v = (int)(i % R3);
  int b = (int)(i / ((size_t)C * R3));
  vox[i] /= fmaxf(cnt[(size_t)b * R3 + v], 1.f);
}

// im2col for one batch element, over column chunk [n0, n0+ncols)
__global__ void k_im2col(const float* __restrict__ vox, f16* __restrict__ col,
                         int C, int r, int n0, int ncols) {
  size_t total = (size_t)C * 27 * ncols;
  size_t i = (size_t)blockIdx.x * 256 + threadIdx.x;
  if (i >= total) return;
  int ncol = (int)(i % ncols);
  int row  = (int)(i / ncols);
  int cc = row / 27, kk = row % 27;
  int n = n0 + ncol;
  int r2 = r * r;
  int x = n / r2, y = (n / r) % r, z = n % r;
  int xx = x + kk / 9 - 1;
  int yy = y + (kk / 3) % 3 - 1;
  int zz = z + kk % 3 - 1;
  f16 v = (f16)0;
  if (xx >= 0 && xx < r && yy >= 0 && yy < r && zz >= 0 && zz < r)
    v = (f16)vox[(size_t)cc * (r2 * r) + (size_t)(xx * r + yy) * r + zz];
  col[(size_t)row * ncols + ncol] = v;
}

__global__ void k_devox_add(const float* __restrict__ vox, const float* __restrict__ nc,
                            float* __restrict__ out, int C, int r) {
  size_t total = (size_t)kB * C * kN;
  size_t i = (size_t)blockIdx.x * 256 + threadIdx.x;
  if (i >= total) return;
  int n = (int)(i % kN);
  int c = (int)((i / kN) % C);
  int b = (int)(i / ((size_t)kN * C));
  int R3 = r * r * r;
  float fx = nc[((size_t)b * 3 + 0) * kN + n];
  float fy = nc[((size_t)b * 3 + 1) * kN + n];
  float fz = nc[((size_t)b * 3 + 2) * kN + n];
  int lx = (int)floorf(fx), ly = (int)floorf(fy), lz = (int)floorf(fz);
  float tx = fx - lx, ty = fy - ly, tz = fz - lz;
  int hx = lx + 1; if (hx > r - 1) hx = r - 1;
  int hy = ly + 1; if (hy > r - 1) hy = r - 1;
  int hz = lz + 1; if (hz > r - 1) hz = r - 1;
  const float* vb = vox + ((size_t)b * C + c) * R3;
  float s = 0.f;
  for (int dx = 0; dx < 2; ++dx) {
    int ix = dx ? hx : lx; float wx = dx ? tx : 1.f - tx;
    for (int dy = 0; dy < 2; ++dy) {
      int iy = dy ? hy : ly; float wy = dy ? ty : 1.f - ty;
      for (int dz = 0; dz < 2; ++dz) {
        int iz = dz ? hz : lz; float wz = dz ? tz : 1.f - tz;
        s += wx * wy * wz * vb[(size_t)(ix * r + iy) * r + iz];
      }
    }
  }
  out[i] += s;
}

// ---------------------------------------------------------------------
//  Misc net pieces
// ---------------------------------------------------------------------
__global__ void k_rowmax(const float* __restrict__ f, float* __restrict__ out, int C, int Nn) {
  int i = blockIdx.x * 256 + threadIdx.x;
  if (i >= kB * C) return;
  const float* row = f + (size_t)i * Nn;
  float m = -3.0e38f;
  for (int n = 0; n < Nn; ++n) m = fmaxf(m, row[n]);
  out[i] = m;
}

__global__ void k_label_feat(const int* __restrict__ label, const float* __restrict__ emb,
                             float* __restrict__ lf0) {
  int i = blockIdx.x * 256 + threadIdx.x;
  if (i >= kB * kNumCat * 3) return;
  int d = i % 3;
  int j = (i / 3) % kNumCat;
  int b = i / (3 * kNumCat);
  int cls = label[b * kNumCat + j];
  lf0[i] = (cls == 0) ? 0.f : emb[cls * 3 + d];   // emb.at[0].set(0)
}

// y0 f16 : (P, LATENT+2, B*G);  rows 0..1 = rand_grid, rows 2.. = features bcast
__global__ void k_build_y0(const float* __restrict__ rg, const float* __restrict__ feat,
                           f16* __restrict__ y0) {
  size_t total = (size_t)kP * (kLatent + 2) * (kB * kG);
  size_t i = (size_t)blockIdx.x * 256 + threadIdx.x;
  if (i >= total) return;
  int col = (int)(i % (kB * kG));
  int row = (int)((i / (kB * kG)) % (kLatent + 2));
  int p   = (int)(i / ((size_t)(kB * kG) * (kLatent + 2)));
  int b = col / kG, g = col % kG;
  float v;
  if (row < 2) v = rg[((size_t)(p * kB + b) * 2 + row) * kG + g];
  else         v = feat[(size_t)b * kLatent + (row - 2)];
  y0[i] = (f16)v;
}

// y4 (P,6,B*G) -> coarse (B, P*G, 6)
__global__ void k_extract_coarse(const float* __restrict__ y4, float* __restrict__ coarse) {
  size_t total = (size_t)kB * (kP * kG) * 6;
  size_t i = (size_t)blockIdx.x * 256 + threadIdx.x;
  if (i >= total) return;
  int c  = (int)(i % 6);
  int pg = (int)((i / 6) % (kP * kG));
  int b  = (int)(i / (6 * (size_t)kP * kG));
  int p = pg / kG, g = pg % kG;
  coarse[i] = y4[((size_t)p * 6 + c) * (kB * kG) + b * kG + g];
}

// Prim's MST per group of G points; one WG / group
__global__ void __launch_bounds__(128)
k_mst(const float* __restrict__ coarse, float* __restrict__ edges) {
  int grp = blockIdx.x;              // b*P + p
  int t   = threadIdx.x;             // 0..127
  __shared__ float px[kG], py[kG], pz[kG], mind[kG], edge[kG];
  __shared__ float red[kG];
  __shared__ int   redi[kG];
  __shared__ unsigned char intree[kG];
  const float* pr = coarse + ((size_t)grp * kG + t) * 6;
  px[t] = pr[0]; py[t] = pr[1]; pz[t] = pr[2];
  edge[t] = 0.f;
  intree[t] = (t == 0) ? 1 : 0;
  __syncthreads();
  float dx = px[t] - px[0], dy = py[t] - py[0], dz = pz[t] - pz[0];
  mind[t] = (t == 0) ? 0.f : sqrtf(dx * dx + dy * dy + dz * dz + 1e-12f);
  __syncthreads();
  for (int it = 0; it < kG - 1; ++it) {
    red[t]  = intree[t] ? 3.0e38f : mind[t];
    redi[t] = t;
    __syncthreads();
    for (int off = 64; off; off >>= 1) {
      if (t < off) {
        if (red[t + off] < red[t] || (red[t + off] == red[t] && redi[t + off] < redi[t])) {
          red[t] = red[t + off]; redi[t] = redi[t + off];
        }
      }
      __syncthreads();
    }
    int j = redi[0];
    if (t == j) { edge[t] = mind[t]; intree[t] = 1; }
    __syncthreads();
    float ax = px[t] - px[j], ay = py[t] - py[j], az = pz[t] - pz[j];
    mind[t] = fminf(mind[t], sqrtf(ax * ax + ay * ay + az * az + 1e-12f));
    __syncthreads();
  }
  edges[(size_t)grp * kG + t] = edge[t];
}

__global__ void k_mean_mst(const float* __restrict__ edges, float* __restrict__ mm) {
  int b = blockIdx.x, t = threadIdx.x;
  __shared__ float red[256];
  float acc = 0.f;
  for (int i = t; i < kP * kG; i += 256) acc += edges[(size_t)b * kP * kG + i];
  red[t] = acc; __syncthreads();
  for (int off = 128; off; off >>= 1) { if (t < off) red[t] += red[t + off]; __syncthreads(); }
  if (t == 0) mm[b] = red[0] / (float)(kP * (kG - 1));
}

__global__ void k_exp_loss(const float* __restrict__ edges, const float* __restrict__ mm,
                           float* __restrict__ outp) {
  __shared__ float red[256];
  int t = threadIdx.x;
  float acc = 0.f;
  for (int i = t; i < kB * kP * kG; i += 256) {
    int b = i / (kP * kG);
    float e = edges[i];
    acc += (e > kAlpha * mm[b]) ? e : 0.f;
  }
  red[t] = acc; __syncthreads();
  for (int off = 128; off; off >>= 1) { if (t < off) red[t] += red[t + off]; __syncthreads(); }
  if (t == 0) outp[0] = red[0] / (float)(kB * kP * kG);
}

__global__ void k_concat_dense(const float* __restrict__ pt, const float* __restrict__ coarse,
                               float* __restrict__ dense) {
  size_t total = (size_t)kB * kMdsM * 6;
  size_t i = (size_t)blockIdx.x * 256 + threadIdx.x;
  if (i >= total) return;
  int c = (int)(i % 6);
  int n = (int)((i / 6) % kMdsM);
  int b = (int)(i / (6 * (size_t)kMdsM));
  dense[i] = (n < kN) ? pt[((size_t)b * kN + n) * 6 + c]
                      : coarse[((size_t)b * (kP * kG) + (n - kN)) * 6 + c];
}

// min-density sampling; one WG per batch item; state LDS-resident (~60 KB)
__global__ void __launch_bounds__(1024)
k_mds(const float* __restrict__ dense, const float* __restrict__ mm, int* __restrict__ idx) {
  int b = blockIdx.x, t = threadIdx.x;
  __shared__ float X[kMdsM], Y[kMdsM], Z[kMdsM], dens[kMdsM];
  __shared__ unsigned char picked[kMdsM];
  __shared__ float red[1024];
  __shared__ int   redi[1024];
  for (int i = t; i < kMdsM; i += 1024) {
    const float* pr = dense + ((size_t)b * kMdsM + i) * 6;
    X[i] = pr[0]; Y[i] = pr[1]; Z[i] = pr[2];
    dens[i] = 0.f; picked[i] = 0;
  }
  float sgm = mm[b];
  float sig2 = 2.f * sgm * sgm + 1e-12f;
  __syncthreads();
  for (int s = 0; s < kMdsN; ++s) {
    float best = 3.0e38f; int bi = kMdsM;
    for (int i = t; i < kMdsM; i += 1024) {
      float v = picked[i] ? 3.0e38f : dens[i];
      if (v < best) { best = v; bi = i; }
    }
    red[t] = best; redi[t] = bi;
    __syncthreads();
    for (int off = 512; off; off >>= 1) {
      if (t < off) {
        if (red[t + off] < red[t] || (red[t + off] == red[t] && redi[t + off] < redi[t])) {
          red[t] = red[t + off]; redi[t] = redi[t + off];
        }
      }
      __syncthreads();
    }
    int j = redi[0];
    float xj = X[j], yj = Y[j], zj = Z[j];
    __syncthreads();
    for (int i = t; i < kMdsM; i += 1024) {
      float dx = X[i] - xj, dy = Y[i] - yj, dz = Z[i] - zj;
      dens[i] += expf(-(dx * dx + dy * dy + dz * dz) / sig2);
    }
    if (t == 0) { picked[j] = 1; idx[(size_t)b * kMdsN + s] = j; }
    __syncthreads();
  }
}

__global__ void k_gather_rows(const float* __restrict__ dense, const int* __restrict__ idx,
                              float* __restrict__ out) {
  size_t total = (size_t)kB * kMdsN * 6;
  size_t i = (size_t)blockIdx.x * 256 + threadIdx.x;
  if (i >= total) return;
  int c = (int)(i % 6);
  int n = (int)((i / 6) % kMdsN);
  int b = (int)(i / (6 * (size_t)kMdsN));
  int j = idx[(size_t)b * kMdsN + n];
  out[i] = dense[((size_t)b * kMdsM + j) * 6 + c];
}

// write layer output (B,C,N) into concat buffer (B,1008,N) at channel coff
__global__ void k_copy_concat(const float* __restrict__ f, float* __restrict__ oc,
                              int C, int coff) {
  size_t total = (size_t)kB * C * kN;
  size_t i = (size_t)blockIdx.x * 256 + threadIdx.x;
  if (i >= total) return;
  int n = (int)(i % kN);
  int c = (int)((i / kN) % C);
  int b = (int)(i / ((size_t)kN * C));
  oc[((size_t)b * 1008 + coff + c) * kN + n] = f[i];
}

// dense_out (B,4096,6): [0,2048) from gathered dense, [2048,4096) from o3 (B,6,2048)
__global__ void k_final_dense(const float* __restrict__ dense_s, const float* __restrict__ o3,
                              float* __restrict__ out) {
  size_t total = (size_t)kB * (2 * kMdsN) * 6;
  size_t i = (size_t)blockIdx.x * 256 + threadIdx.x;
  if (i >= total) return;
  int c = (int)(i % 6);
  int n = (int)((i / 6) % (2 * kMdsN));
  int b = (int)(i / (6 * (size_t)(2 * kMdsN)));
  out[i] = (n < kMdsN) ? dense_s[((size_t)b * kMdsN + n) * 6 + c]
                       : o3[((size_t)b * 6 + c) * kMdsN + (n - kMdsN)];
}

// =====================================================================
//  Host orchestration
// =====================================================================
extern "C" void kernel_launch(void* const* d_in, const int* in_sizes, int n_in,
                              void* d_out, int out_size, void* d_ws, size_t ws_size,
                              hipStream_t stream)
{
  (void)in_sizes; (void)n_in; (void)out_size; (void)ws_size;

  // ----- input unpack (JAX pytree order: dict keys sorted) -----
  const int*   label     = (const int*)  d_in[0];
  const float* pt        = (const float*)d_in[1];
  const float* rand_grid = (const float*)d_in[2];
  const float* dec_w[4]  = { (const float*)d_in[3], (const float*)d_in[4],
                             (const float*)d_in[5], (const float*)d_in[6] };
  const float* emb       = (const float*)d_in[7];
  const float *enc_pw[3], *enc_v1[3], *enc_v2[3];
  for (int l = 0; l < 3; ++l) {
    enc_pw[l] = (const float*)d_in[8 + 3 * l];
    enc_v1[l] = (const float*)d_in[9 + 3 * l];
    enc_v2[l] = (const float*)d_in[10 + 3 * l];
  }
  const float* lab_b1 = (const float*)d_in[17];
  const float* lab_b2 = (const float*)d_in[18];
  const float* lab_w1 = (const float*)d_in[19];
  const float* lab_w2 = (const float*)d_in[20];
  const float* lin_b  = (const float*)d_in[21];
  const float* lin_w  = (const float*)d_in[22];
  const float* res_mlp_w[3] = { (const float*)d_in[23], (const float*)d_in[24],
                                (const float*)d_in[25] };
  const float *res_pw[6], *res_v1[6], *res_v2[6];
  for (int l = 0; l < 6; ++l) {
    res_pw[l] = (const float*)d_in[26 + 3 * l];
    res_v1[l] = (const float*)d_in[27 + 3 * l];
    res_v2[l] = (const float*)d_in[28 + 3 * l];
  }

  // ----- workspace carve -----
  char* wsbase = (char*)d_ws;
  size_t wsoff = 0;
  auto alloc = [&](size_t bytes) -> void* {
    void* p = wsbase + wsoff;
    wsoff += (bytes + 255) & ~(size_t)255;
    return p;
  };
  float* ptT    = (float*)alloc((size_t)kB * 6 * kN * 4);
  float* meanv  = (float*)alloc(kB * 3 * 4);
  float* scalev = (float*)alloc(kB * 4);
  float* nc     = (float*)alloc((size_t)kB * 3 * kN * 4);
  int*   flat   = (int*)  alloc((size_t)kB * kN * 4);
  float* vox0   = (float*)alloc((size_t)kB * 256 * 4096 * 4);  // conv input voxels
  float* vox1   = (float*)alloc((size_t)kB * 512 * 4096 * 4);  // conv1 out
  float* vox2   = (float*)alloc((size_t)kB * 512 * 4096 * 4);  // conv2 out (also o3 later)
  float* cnt    = (float*)alloc((size_t)kB * 4096 * 4);
  f16*   colh   = (f16*)  alloc((size_t)13824 * 1024 * 2);     // im2col chunk
  f16*   wh     = (f16*)  alloc((size_t)kP * 1026 * 1056 * 2); // f16 weights (lda-padded)
  f16*   xh     = (f16*)  alloc((size_t)kB * 1008 * kN * 2);   // f16 activations
  float* fA     = (float*)alloc((size_t)kB * 512 * kN * 4);
  float* fB     = (float*)alloc((size_t)kB * 512 * kN * 4);
  float* ptf    = (float*)alloc((size_t)kB * 512 * 4);
  float* feat   = (float*)alloc((size_t)kB * kLatent * 4);
  float* lf0    = (float*)alloc((size_t)kB * 144 * 4);
  float* lf1    = (float*)alloc((size_t)kB * 256 * 4);
  f16*   sh     = (f16*)  alloc((size_t)kB * kLatent * 2);     // small f16 A scratch
  f16*   yH     = (f16*)  alloc((size_t)kP * 1026 * (kB * kG) * 2);
  float* yF     = (float*)alloc((size_t)kP * 1026 * (kB * kG) * 4);
  float* coarse = (float*)alloc((size_t)kB * kP * kG * 6 * 4);
  float* edges  = (float*)alloc((size_t)kB * kP * kG * 4);
  float* mm     = (float*)alloc((size_t)kB * 4);
  float* dense  = (float*)alloc((size_t)kB * kMdsM * 6 * 4);
  int*   idx    = (int*)  alloc((size_t)kB * kMdsN * 4);
  float* dense_s= (float*)alloc((size_t)kB * kMdsN * 6 * 4);
  float* featsT = (float*)alloc((size_t)kB * 6 * kMdsN * 4);
  float* oc     = (float*)alloc((size_t)kB * 1008 * kN * 4);

  auto nb = [](size_t tot) { return dim3((unsigned)((tot + 255) / 256)); };
  auto pad32 = [](int k) { return (k + 31) & ~31; };
  auto gemm = [&](const f16* A, const f16* Bm, float* C, const float* bias,
                  int M, int Nn, int K, int lda, int ldc, int act) {
    dim3 grid((unsigned)((Nn + 63) / 64), (unsigned)((M + 127) / 128));
    k_gemm_wmma<<<grid, 256, 0, stream>>>(A, Bm, C, bias, M, Nn, K, lda, ldc, act);
  };
  auto cast16 = [&](const float* in, f16* outp, size_t n) {
    k_cast_f32_f16<<<nb(n), 256, 0, stream>>>(in, outp, n);
  };
  auto castPad = [&](const float* in, f16* outp, int M, int K, int lda) {
    k_cast_pad<<<nb((size_t)M * lda), 256, 0, stream>>>(in, outp, M, K, lda);
  };
  auto castT = [&](const float* in, f16* outp, int M, int K) {
    dim3 g((unsigned)((K + 15) / 16), (unsigned)((M + 15) / 16));
    k_cast_transpose_f32_f16<<<g, dim3(16, 16), 0, stream>>>(in, outp, M, K);
  };
  auto zero = [&](float* p, size_t n) { k_zero_f32<<<nb(n), 256, 0, stream>>>(p, n); };

  // ----- shared PVConv pipeline -----
  auto pvconv = [&](const float* fin, const float* coordsp, int cstride,
                    const float* w1, const float* w2, const float* pw,
                    int Cin, int Cout, int r, float* fout) {
    int R3 = r * r * r;
    k_voxel_stats <<<kB, 256, 0, stream>>>(coordsp, cstride, meanv, scalev, kN);
    k_voxel_coords<<<nb((size_t)kB * kN), 256, 0, stream>>>(coordsp, cstride, meanv, scalev,
                                                            nc, flat, kN, r);
    zero(vox0, (size_t)kB * Cin * R3);
    zero(cnt, (size_t)kB * R3);
    k_voxel_scatter<<<nb((size_t)kB * Cin * kN), 256, 0, stream>>>(fin, flat, vox0, cnt, Cin, kN, R3);
    k_voxel_div<<<nb((size_t)kB * Cin * R3), 256, 0, stream>>>(vox0, cnt, Cin, R3);

    int chunk = (R3 > 1024) ? 1024 : R3;
    // conv1: vox0 -> vox1 (leaky relu)
    {
      int K1 = Cin * 27, lda1 = pad32(K1);
      castPad(w1, wh, Cout, K1, lda1);
      for (int b = 0; b < kB; ++b)
        for (int n0 = 0; n0 < R3; n0 += chunk) {
          int ncols = (R3 - n0 < chunk) ? (R3 - n0) : chunk;
          k_im2col<<<nb((size_t)K1 * ncols), 256, 0, stream>>>(
              vox0 + (size_t)b * Cin * R3, colh, Cin, r, n0, ncols);
          gemm(wh, colh, vox1 + (size_t)b * Cout * R3 + n0, nullptr,
               Cout, ncols, K1, lda1, R3, ACT_LEAKY);
        }
    }
    // conv2: vox1 -> vox2 (leaky relu)
    {
      int K2 = Cout * 27, lda2 = pad32(K2);
      castPad(w2, wh, Cout, K2, lda2);
      for (int b = 0; b < kB; ++b)
        for (int n0 = 0; n0 < R3; n0 += chunk) {
          int ncols = (R3 - n0 < chunk) ? (R3 - n0) : chunk;
          k_im2col<<<nb((size_t)K2 * ncols), 256, 0, stream>>>(
              vox1 + (size_t)b * Cout * R3, colh, Cout, r, n0, ncols);
          gemm(wh, colh, vox2 + (size_t)b * Cout * R3 + n0, nullptr,
               Cout, ncols, K2, lda2, R3, ACT_LEAKY);
        }
    }
    // pointwise: fout = relu(pw @ fin), then += trilinear(vox2)
    {
      int ldap = pad32(Cin);
      cast16(fin, xh, (size_t)kB * Cin * kN);
      castPad(pw, wh, Cout, Cin, ldap);
      for (int b = 0; b < kB; ++b)
        gemm(wh, xh + (size_t)b * Cin * kN, fout + (size_t)b * Cout * kN, nullptr,
             Cout, kN, Cin, ldap, kN, ACT_RELU);
    }
    k_devox_add<<<nb((size_t)kB * Cout * kN), 256, 0, stream>>>(vox2, nc, fout, Cout, r);
  };

  // ================= encoder =================
  k_transpose_bnc_bcn<<<nb((size_t)kB * kN * 6), 256, 0, stream>>>(pt, ptT, kN, 6);
  pvconv(ptT, ptT, 6 * kN, enc_v1[0], enc_v2[0], enc_pw[0], 6,   256, 16, fA);
  pvconv(fA,  ptT, 6 * kN, enc_v1[1], enc_v2[1], enc_pw[1], 256, 512, 8,  fB);
  pvconv(fB,  ptT, 6 * kN, enc_v1[2], enc_v2[2], enc_pw[2], 512, 512, 4,  fA);
  k_rowmax<<<nb((size_t)kB * 512), 256, 0, stream>>>(fA, ptf, 512, kN);

  // pt_feat = elu(ptf @ lin_w.T + lin_b)  -> features[:, 0:512]
  castPad(ptf, sh, kB, 512, 512);
  castT(lin_w, wh, 512, 512);
  gemm(sh, wh, feat, lin_b, kB, 512, 512, 512, kLatent, ACT_ELU);

  // label path -> features[:, 512:1024]
  k_label_feat<<<nb((size_t)kB * kNumCat * 3), 256, 0, stream>>>(label, emb, lf0);
  castPad(lf0, sh, kB, 144, 160);
  castT(lab_w1, wh, 256, 144);
  gemm(sh, wh, lf1, lab_b1, kB, 256, 144, 160, 256, ACT_NONE);
  castPad(lf1, sh, kB, 256, 256);
  castT(lab_w2, wh, 512, 256);
  gemm(sh, wh, feat + 512, lab_b2, kB, 512, 256, 256, kLatent, ACT_NONE);

  // ================= decoder =================
  k_build_y0<<<nb((size_t)kP * 1026 * (kB * kG)), 256, 0, stream>>>(rand_grid, feat, yH);
  {
    const int dims[5] = {1026, 1026, 513, 256, 6};
    for (int l = 0; l < 4; ++l) {
      int cI = dims[l], cO = dims[l + 1];
      int ldaI = pad32(cI);
      castPad(dec_w[l], wh, kP * cO, cI, ldaI);   // (P*cO) rows, lda-padded
      for (int p = 0; p < kP; ++p)
        gemm(wh + (size_t)p * cO * ldaI, yH + (size_t)p * cI * (kB * kG),
             yF + (size_t)p * cO * (kB * kG), nullptr,
             cO, kB * kG, cI, ldaI, kB * kG, ACT_ELU);
      if (l < 3) cast16(yF, yH, (size_t)kP * cO * (kB * kG));
    }
  }
  k_extract_coarse<<<nb((size_t)kB * kP * kG * 6), 256, 0, stream>>>(yF, coarse);

  // ================= expansion penalty =================
  k_mst<<<kB * kP, 128, 0, stream>>>(coarse, edges);
  k_mean_mst<<<kB, 256, 0, stream>>>(edges, mm);

  // ================= MDS + gather =================
  k_concat_dense<<<nb((size_t)kB * kMdsM * 6), 256, 0, stream>>>(pt, coarse, dense);
  k_mds<<<kB, 1024, 0, stream>>>(dense, mm, idx);
  k_gather_rows<<<nb((size_t)kB * kMdsN * 6), 256, 0, stream>>>(dense, idx, dense_s);
  k_transpose_bnc_bcn<<<nb((size_t)kB * kMdsN * 6), 256, 0, stream>>>(dense_s, featsT, kMdsN, 6);

  // ================= residual PVConv stack =================
  {
    const int rc_in[6]  = {6, 16, 32, 64, 128, 256};
    const int rc_out[6] = {16, 32, 64, 128, 256, 512};
    const int rr[6]     = {4, 4, 8, 8, 16, 16};
    const float* cur = featsT;
    float* bufs[2] = {fA, fB};
    int coff = 0;
    for (int l = 0; l < 6; ++l) {
      float* outp = bufs[l & 1];
      pvconv(cur, featsT, 6 * kN, res_v1[l], res_v2[l], res_pw[l],
             rc_in[l], rc_out[l], rr[l], outp);
      k_copy_concat<<<nb((size_t)kB * rc_out[l] * kN), 256, 0, stream>>>(outp, oc, rc_out[l], coff);
      coff += rc_out[l];
      cur = outp;
    }
  }

  // ================= residual MLP =================
  cast16(oc, xh, (size_t)kB * 1008 * kN);
  castPad(res_mlp_w[0], wh, 512, 1008, 1024);
  for (int b = 0; b < kB; ++b)
    gemm(wh, xh + (size_t)b * 1008 * kN, fA + (size_t)b * 512 * kN, nullptr,
         512, kN, 1008, 1024, kN, ACT_RELU);
  cast16(fA, xh, (size_t)kB * 512 * kN);
  castPad(res_mlp_w[1], wh, 256, 512, 512);
  for (int b = 0; b < kB; ++b)
    gemm(wh, xh + (size_t)b * 512 * kN, fB + (size_t)b * 256 * kN, nullptr,
         256, kN, 512, 512, kN, ACT_RELU);
  cast16(fB, xh, (size_t)kB * 256 * kN);
  castPad(res_mlp_w[2], wh, 6, 256, 256);
  float* o3 = vox2;   // reuse (B,6,2048)
  for (int b = 0; b < kB; ++b)
    gemm(wh, xh + (size_t)b * 256 * kN, o3 + (size_t)b * 6 * kN, nullptr,
         6, kN, 256, 256, kN, ACT_RELU);

  // ================= outputs =================
  const size_t coarse_sz = (size_t)kB * kP * kG * 6;         // 24576
  const size_t dense_sz  = (size_t)kB * (2 * kMdsN) * 6;     // 98304
  k_copy_f32<<<nb(coarse_sz), 256, 0, stream>>>(coarse, (float*)d_out, coarse_sz);
  k_final_dense<<<nb(dense_sz), 256, 0, stream>>>(dense_s, o3, (float*)d_out + coarse_sz);
  k_exp_loss<<<1, 256, 0, stream>>>(edges, mm, (float*)d_out + coarse_sz + dense_sz);
}
